// MultiheadSelfAttention_59261958750492
// MI455X (gfx1250) — compile-verified, matrix operand order checked by reading, not run on hardware
//
#include <hip/hip_runtime.h>
#include <math.h>

// Problem constants (from reference): N=4, S=2048, D=1024, H=16, DH=64
#define NBATCH 4
#define S_LEN  2048
#define D_EMB  1024
#define NHEAD  16
#define DHEAD  64

// Workspace layout (bf16 as ushort): Q | K | V, each N*H*S*DH elements
// = 3 * 8,388,608 * 2 bytes = 48 MB required in d_ws.

typedef __attribute__((ext_vector_type(16))) __bf16 v16bf;
typedef __attribute__((ext_vector_type(8)))  float  v8f;
typedef __attribute__((ext_vector_type(4)))  unsigned int v4u;
typedef __attribute__((ext_vector_type(8)))  int v8i;
typedef __attribute__((ext_vector_type(4)))  int v4i;

union Frag { v16bf v; unsigned int u[8]; };

// f32 -> bf16, round-to-nearest-even
__device__ __forceinline__ unsigned short f2bf(float f) {
  unsigned int u = __float_as_uint(f);
  u += 0x7FFFu + ((u >> 16) & 1u);
  return (unsigned short)(u >> 16);
}

// 16-bit A/B operand K-index map (ISA 7.12.2):
// VGPR v in 0..3 -> K = 2v        + 8*half
// VGPR v in 4..7 -> K = 16+2(v-4) + 8*half   (half = lane/16)
__device__ __forceinline__ int kmap(int v, int half) {
  return (v < 4) ? (2 * v + 8 * half) : (16 + 2 * (v - 4) + 8 * half);
}

#define WMMA_BF16(A, B, C) \
  __builtin_amdgcn_wmma_f32_16x16x32_bf16(false, (A), false, (B), (short)0, (C), false, false)

// ---------------------------------------------------------------------------
// TDM: DMA a 64x64 bf16 tile (row stride 64 elems in global) into LDS with an
// 8-byte pad after every 128-byte row => LDS row stride 136 B = 68 shorts.
// D# per CDNA5 ISA (group0: count/lds/global/type,
// group1: data_size=2B, pad_interval=4 (32 dwords), pad_amount=1 (2 dwords),
// tensor_dim0/1 = 64, tile_dim0/1 = 64, tensor_dim0_stride = 64).
// Toolchain here is the clang-23 lane: 6-arg builtin
//   (uint32x4 g0, int32x8 g1, int32x4, int32x4, int32x8, i32 cpol)
// ---------------------------------------------------------------------------
__device__ __forceinline__ void tdm_load_tile_bf16(const unsigned short* gsrc,
                                                   unsigned lds_byte_off) {
  unsigned long long ga = (unsigned long long)(size_t)gsrc;
  v4u g0 = { 1u,                                          // count=1, user mode
             lds_byte_off,                                // lds_addr
             (unsigned)(ga & 0xFFFFFFFFu),                // global_addr[31:0]
             (unsigned)((ga >> 32) & 0x01FFFFFFu) | (2u << 30) };  // [56:32] | type=2
  v8i g1 = { (int)((1u << 16) | (1u << 20) | (4u << 22) | (1u << 25)),
             //     data_size=2B  pad_en     pad_int=32dw  pad_amt=2dw ; wg_mask=0
             (int)(64u << 16),   // tensor_dim0[15:0] in high half
             (int)(64u << 16),   // tensor_dim0[31:16]=0 | tensor_dim1[15:0]
             (int)(64u << 16),   // tensor_dim1[31:16]=0 | tile_dim0=64
             (int)64,            // tile_dim1=64 | tile_dim2=0
             (int)64,            // tensor_dim0_stride[31:0]=64
             0,                  // stride0[47:32]=0 | stride1[15:0]=0
             0 };                // stride1[47:16]=0
  v4i z4 = {0, 0, 0, 0};
  v8i z8 = {0, 0, 0, 0, 0, 0, 0, 0};
  __builtin_amdgcn_tensor_load_to_lds(g0, g1, z4, z4, z8, 0);
}

// ---------------------------------------------------------------------------
// Kernel 1: per-head QKV projection.  grid = (S/64, N*H), block = 128 (4 waves)
// q[s,e] = sum_d x[s, h*64+d] * W[h][d][e] + b[h][e]   for W in {Wq,Wk,Wv}
// ---------------------------------------------------------------------------
__global__ __launch_bounds__(128)
void qkv_proj_kernel(const float* __restrict__ x,
                     const float* __restrict__ Wq, const float* __restrict__ bq,
                     const float* __restrict__ Wk, const float* __restrict__ bk,
                     const float* __restrict__ Wv, const float* __restrict__ bv,
                     unsigned short* __restrict__ Qb,
                     unsigned short* __restrict__ Kb,
                     unsigned short* __restrict__ Vb) {
  constexpr int STR = 68;  // padded LDS row stride (shorts)
  __shared__ __align__(16) unsigned short lds_X[64 * STR];       // X[s][d] bf16
  __shared__ __align__(16) unsigned short lds_WT[3][64 * STR];   // W^T[e][d] bf16

  const int tid  = threadIdx.x;
  const int wave = tid >> 5, lane = tid & 31;
  const int half = lane >> 4, l16 = lane & 15;
  const int sb = blockIdx.x * 64;
  const int nh = blockIdx.y;
  const int n = nh >> 4, h = nh & 15;

  // Stage x tile (64 rows x 64 cols of this head's slice), converted to bf16.
  const float* xg = x + (size_t)n * S_LEN * D_EMB + (size_t)h * DHEAD;
  for (int idx = tid; idx < 64 * 64; idx += 128) {
    int s = idx >> 6, d = idx & 63;
    lds_X[s * STR + d] = f2bf(xg[(size_t)(sb + s) * D_EMB + d]);
  }
  // Stage W^T (so B-operand pairs along d are contiguous 32-bit LDS loads).
  const float* wsrc[3] = { Wq + (size_t)h * DHEAD * DHEAD,
                           Wk + (size_t)h * DHEAD * DHEAD,
                           Wv + (size_t)h * DHEAD * DHEAD };
#pragma unroll
  for (int w = 0; w < 3; ++w)
    for (int idx = tid; idx < 64 * 64; idx += 128) {
      int d = idx >> 6, e = idx & 63;
      lds_WT[w][e * STR + d] = f2bf(wsrc[w][d * DHEAD + e]);
    }
  __syncthreads();

  // A-operands: this wave's 16 x-rows, two K-chunks (d 0..31, 32..63).
  Frag a0, a1;
  {
    const int srow = wave * 16 + l16;
#pragma unroll
    for (int v = 0; v < 8; ++v) {
      int d = kmap(v, half);
      a0.u[v] = *(const unsigned int*)&lds_X[srow * STR + d];
      a1.u[v] = *(const unsigned int*)&lds_X[srow * STR + 32 + d];
    }
  }

  const float* bias[3] = { bq + h * DHEAD, bk + h * DHEAD, bv + h * DHEAD };
  unsigned short* dst[3] = { Qb, Kb, Vb };
  const size_t obase = (size_t)nh * S_LEN * DHEAD;

#pragma unroll
  for (int w = 0; w < 3; ++w) {
#pragma unroll
    for (int t = 0; t < 4; ++t) {
      Frag b0, b1;
      const int e = t * 16 + l16;
#pragma unroll
      for (int v = 0; v < 8; ++v) {
        int d = kmap(v, half);
        b0.u[v] = *(const unsigned int*)&lds_WT[w][e * STR + d];
        b1.u[v] = *(const unsigned int*)&lds_WT[w][e * STR + 32 + d];
      }
      v8f acc = {0.f, 0.f, 0.f, 0.f, 0.f, 0.f, 0.f, 0.f};
      acc = WMMA_BF16(a0.v, b0.v, acc);
      acc = WMMA_BF16(a1.v, b1.v, acc);
      const float bb = bias[w][e];
      unsigned short* od = dst[w] + obase;
#pragma unroll
      for (int v = 0; v < 8; ++v) {
        int srow = sb + wave * 16 + v + 8 * half;  // C layout: M = v + 8*half
        od[(size_t)srow * DHEAD + e] = f2bf(acc[v] + bb);
      }
    }
  }
}

// ---------------------------------------------------------------------------
// Kernel 2: flash attention per (n,h).  grid = (S/64, N*H), block = 128.
// Each wave owns 16 q-rows; block streams 64-wide K/V tiles through LDS.
// K tile is DMA'd by the Tensor Data Mover (async, LDS row-padding in D#);
// V tile is transposed during staging for the P.V B-operands.
// ---------------------------------------------------------------------------
__global__ __launch_bounds__(128)
void flash_attn_kernel(const unsigned short* __restrict__ Qb,
                       const unsigned short* __restrict__ Kb,
                       const unsigned short* __restrict__ Vb,
                       float* __restrict__ out) {
  constexpr int STR = 68;
  __shared__ __align__(16) unsigned short lds_K[64 * STR];   // K[k][d] bf16 (TDM)
  __shared__ __align__(16) unsigned short lds_VT[64 * STR];  // V^T[d][k] bf16
  __shared__ __align__(16) unsigned short lds_P[4][16 * 64]; // per-wave P slab

  const int tid  = threadIdx.x;
  const int wave = tid >> 5, lane = tid & 31;
  const int half = lane >> 4, l16 = lane & 15;
  const int nh = blockIdx.y;
  const int n = nh >> 4, h = nh & 15;
  const size_t base = (size_t)nh * S_LEN * DHEAD;
  const unsigned short* Qg = Qb + base;
  const unsigned short* Kg = Kb + base;
  const unsigned short* Vg = Vb + base;
  const int q0 = blockIdx.x * 64 + wave * 16;

  // LDS byte offset of the K tile (flat LDS addr truncates to LDS offset).
  const unsigned ldsK_off = (unsigned)(size_t)&lds_K[0];

  // Q A-operands directly from global (rows contiguous along d).
  Frag qa0, qa1;
#pragma unroll
  for (int v = 0; v < 8; ++v) {
    int d = kmap(v, half);
    const unsigned short* qr = Qg + (size_t)(q0 + l16) * DHEAD;
    qa0.u[v] = *(const unsigned int*)(qr + d);
    qa1.u[v] = *(const unsigned int*)(qr + 32 + d);
  }

  v8f o[4];
  float m_run[8], l_run[8];
#pragma unroll
  for (int t = 0; t < 4; ++t)
#pragma unroll
    for (int v = 0; v < 8; ++v) o[t][v] = 0.f;
#pragma unroll
  for (int v = 0; v < 8; ++v) { m_run[v] = -__builtin_inff(); l_run[v] = 0.f; }

  for (int j = 0; j < S_LEN / 64; ++j) {
    const int kb = j * 64;
    __syncthreads();  // previous iteration's LDS reads complete

    // Kick off the TDM for the K tile (one wave issues; EXEC is ignored).
    if (wave == 0) tdm_load_tile_bf16(Kg + (size_t)kb * DHEAD, ldsK_off);

    // Meanwhile all threads stage V transposed for the P.V B-operands.
    for (int idx = tid; idx < 64 * 64; idx += 128) {
      int k = idx >> 6, d = idx & 63;
      lds_VT[d * STR + k] = Vg[(size_t)(kb + k) * DHEAD + d];
    }
    if (wave == 0) __builtin_amdgcn_s_wait_tensorcnt(0);
    __syncthreads();  // K tile (TDM) + V^T visible to all waves

    // scores: S(16q x 64k) = Q(16x64) . K^T, contraction over d in 2 chunks
    v8f sc[4];
#pragma unroll
    for (int t = 0; t < 4; ++t) {
      Frag b0, b1;
      const int kc = t * 16 + l16;  // B lane -> N = k column
#pragma unroll
      for (int v = 0; v < 8; ++v) {
        int d = kmap(v, half);
        b0.u[v] = *(const unsigned int*)&lds_K[kc * STR + d];
        b1.u[v] = *(const unsigned int*)&lds_K[kc * STR + 32 + d];
      }
      v8f z = {0.f, 0.f, 0.f, 0.f, 0.f, 0.f, 0.f, 0.f};
      z = WMMA_BF16(qa0.v, b0.v, z);
      sc[t] = WMMA_BF16(qa1.v, b1.v, z);
    }

    // Online softmax: row r = v + 8*half; its 16 cols/tile live across 16 lanes.
#pragma unroll
    for (int v = 0; v < 8; ++v) {
      float mv = -__builtin_inff();
#pragma unroll
      for (int t = 0; t < 4; ++t) { sc[t][v] *= 0.125f; mv = fmaxf(mv, sc[t][v]); }
#pragma unroll
      for (int off = 1; off < 16; off <<= 1) mv = fmaxf(mv, __shfl_xor(mv, off, 32));
      const float mn   = fmaxf(m_run[v], mv);
      const float corr = __expf(m_run[v] - mn);
      m_run[v] = mn;
      l_run[v] *= corr;
#pragma unroll
      for (int t = 0; t < 4; ++t) o[t][v] *= corr;
      float rs = 0.f;
#pragma unroll
      for (int t = 0; t < 4; ++t) {
        float p = __expf(sc[t][v] - mn);
        sc[t][v] = p;
        rs += p;
      }
#pragma unroll
      for (int off = 1; off < 16; off <<= 1) rs += __shfl_xor(rs, off, 32);
      l_run[v] += rs;
    }

    // C-layout -> A-layout via per-wave LDS slab (bf16 P).
    unsigned short* P = lds_P[wave];
#pragma unroll
    for (int t = 0; t < 4; ++t)
#pragma unroll
      for (int v = 0; v < 8; ++v)
        P[(v + 8 * half) * 64 + t * 16 + l16] = f2bf(sc[t][v]);
    // LDS ops from one wave are in-order; explicit wait as belt-and-braces.
    asm volatile("s_wait_dscnt 0" ::: "memory");

    Frag pa0, pa1;
#pragma unroll
    for (int v = 0; v < 8; ++v) {
      int kk = kmap(v, half);
      pa0.u[v] = *(const unsigned int*)&P[l16 * 64 + kk];
      pa1.u[v] = *(const unsigned int*)&P[l16 * 64 + 32 + kk];
    }

    // O(16q x 64d) += P(16x64k) . V(64k x 64d), V^T staged so pairs contiguous.
#pragma unroll
    for (int t = 0; t < 4; ++t) {
      Frag b0, b1;
      const int dc = t * 16 + l16;  // B lane -> N = d column
#pragma unroll
      for (int v = 0; v < 8; ++v) {
        int kk = kmap(v, half);
        b0.u[v] = *(const unsigned int*)&lds_VT[dc * STR + kk];
        b1.u[v] = *(const unsigned int*)&lds_VT[dc * STR + 32 + kk];
      }
      o[t] = WMMA_BF16(pa0.v, b0.v, o[t]);
      o[t] = WMMA_BF16(pa1.v, b1.v, o[t]);
    }
  }

  // Epilogue: out[n][q][h*64 + d] = O / l
  float* og = out + (size_t)n * S_LEN * D_EMB + (size_t)h * DHEAD;
#pragma unroll
  for (int t = 0; t < 4; ++t)
#pragma unroll
    for (int v = 0; v < 8; ++v) {
      int q = q0 + v + 8 * half;
      og[(size_t)q * D_EMB + t * 16 + l16] = o[t][v] / l_run[v];
    }
}

// ---------------------------------------------------------------------------
extern "C" void kernel_launch(void* const* d_in, const int* in_sizes, int n_in,
                              void* d_out, int out_size, void* d_ws, size_t ws_size,
                              hipStream_t stream) {
  const float* x  = (const float*)d_in[0];
  const float* Wq = (const float*)d_in[1];
  const float* bq = (const float*)d_in[2];
  const float* Wk = (const float*)d_in[3];
  const float* bk = (const float*)d_in[4];
  const float* Wv = (const float*)d_in[5];
  const float* bv = (const float*)d_in[6];
  float* out = (float*)d_out;

  const size_t elems = (size_t)NBATCH * NHEAD * S_LEN * DHEAD;  // 8,388,608
  unsigned short* Qb = (unsigned short*)d_ws;
  unsigned short* Kb = Qb + elems;
  unsigned short* Vb = Kb + elems;

  dim3 grid(S_LEN / 64, NBATCH * NHEAD);
  qkv_proj_kernel<<<grid, 128, 0, stream>>>(x, Wq, bq, Wk, bk, Wv, bv, Qb, Kb, Vb);
  flash_attn_kernel<<<grid, 128, 0, stream>>>(Qb, Kb, Vb, out);
}
